// MDL_RNN_mante_4174708212035
// MI455X (gfx1250) — matrix operand order repaired and checked by reading, not compiled, not run on hardware
//
#include <hip/hip_runtime.h>

// ---------------------------------------------------------------------------
// Spiking low-rank RNN scan for MI455X (gfx1250, wave32, WMMA).
// 8 workgroups x 16 wave32s; each WG owns 16 batches (one WMMA N tile).
// All weight A-operands live in VGPRs for the whole 512-step scan; per-(h,b)
// state lives in VGPRs in the WMMA C/D layout; only activations (r, u, x)
// round-trip LDS. 3 barriers per step.
// ---------------------------------------------------------------------------

typedef __attribute__((ext_vector_type(16))) __bf16         v16bf;
typedef __attribute__((ext_vector_type(16))) unsigned short v16us;
typedef __attribute__((ext_vector_type(8)))  unsigned short v8us;
typedef __attribute__((ext_vector_type(8)))  float          v8f;

#define T_STEPS  512
#define BATCH    128
#define INP      32
#define HID      512
#define OUTN     8
#define PRANK    64
#define NB       16    // batches per workgroup (= one WMMA N tile)
#define NTHREADS 512   // 16 wave32s

__device__ __forceinline__ unsigned short f2bf(float f) {
  // round-to-nearest-even fp32 -> bf16
  unsigned int u = __float_as_uint(f);
  u += 0x7FFFu + ((u >> 16) & 1u);
  return (unsigned short)(u >> 16);
}

// A-operand slot map (16-bit A 16x32): lane l, element j
//   row = l & 15,  K = j + 16*(l>>4)
__device__ __forceinline__ int kmapA(int l2, int j) {
  return j + 16 * (l2 >> 4);
}
// B-operand slot map (16-bit B 32x16): lanes 0-15 hold K=0..15, 16-31 K=16..31
__device__ __forceinline__ int kmapB(int l2, int j) {
  return j + 16 * (l2 >> 4);
}

__device__ __forceinline__ v8f wmma_bf16(v16bf a, v16bf b, v8f c) {
  return __builtin_amdgcn_wmma_f32_16x16x32_bf16(
      /*neg_a=*/false, a, /*neg_b=*/false, b,
      /*c_mod=*/(short)0, c, /*reuse_a=*/false, /*reuse_b=*/false);
}

__device__ __forceinline__ v16bf lds_b_tile(const unsigned short* p) {
  return *reinterpret_cast<const v16bf*>(p);  // 2x ds_load_b128
}

__global__ void __launch_bounds__(NTHREADS)
snn_lowrank_wmma_kernel(const float* __restrict__ x,
                        const float* __restrict__ Win,
                        const float* __restrict__ Wout,
                        const float* __restrict__ pin,
                        const float* __restrict__ pout,
                        const float* __restrict__ lvec,
                        float* __restrict__ out) {
  // Only dynamic activations live in LDS (~25 KB).
  __shared__ alignas(16) unsigned short r_t[16 * 512];  // B: r (512x16)
  __shared__ alignas(16) unsigned short u_t[2 * 512];   // B: u (64x16)
  __shared__ alignas(16) unsigned short xs[2][512];     // B: x_t (32x16), dbl buf
  __shared__ float u_f32[64 * 16];                      // K-split reduce buf
  __shared__ float y_f32[8 * 16];                       // y reduce buf

  const int tid   = threadIdx.x;
  const int lane  = tid & 31;
  const int wv    = tid >> 5;   // wave id 0..15
  const int hi    = lane >> 4;  // C/D layout: M = v + 8*hi
  const int ncol  = lane & 15;  // C/D layout: N = batch column
  const int row   = lane & 15;  // A layout row
  const int bbase = blockIdx.x * NB;

  // ---- init: zero LDS, stage x[0] ----
  for (int i = tid; i < 16 * 512; i += NTHREADS) r_t[i] = 0;  // r0 = 0
  u_f32[tid] = 0.f;
  u_f32[tid + 512] = 0.f;
  if (tid < 128) y_f32[tid] = 0.f;
  {
    int l2 = tid & 31, j = tid >> 5;
    int n = l2 & 15, k = kmapB(l2, j);
    xs[0][l2 * 16 + j] = f2bf(x[(size_t)(bbase + n) * INP + k]);
  }

  // ---- one-time: gather invariant A-operand tiles into VGPRs ----
  // wave wv: u-matmul M-tile (wv&3), K-tiles 4*(wv>>2)..+3; I-update M-tiles
  // 2*wv,2*wv+1; y-matmul K-tile wv.
  v16bf a_pout[4], a_pre[2][2], a_win[2], a_wout;
  {
    int mt = wv & 3;
#pragma unroll
    for (int kk = 0; kk < 4; ++kk) {
      int kt = (wv >> 2) * 4 + kk;
      v16us tmp;
#pragma unroll
      for (int j = 0; j < 16; ++j) {
        int h = 32 * kt + kmapA(lane, j);
        tmp[j] = f2bf(pout[h * PRANK + 16 * mt + row]);  // poutT[p][h]
      }
      a_pout[kk] = __builtin_bit_cast(v16bf, tmp);
    }
  }
#pragma unroll
  for (int tt = 0; tt < 2; ++tt) {
    int h = 16 * (2 * wv + tt) + row;
#pragma unroll
    for (int kt = 0; kt < 2; ++kt) {
      v16us tmp;
#pragma unroll
      for (int j = 0; j < 16; ++j) {
        int p = 32 * kt + kmapA(lane, j);
        tmp[j] = f2bf(lvec[p] * pin[h * PRANK + p]);  // Wpre = l*pin
      }
      a_pre[tt][kt] = __builtin_bit_cast(v16bf, tmp);
    }
    v16us tw;
#pragma unroll
    for (int j = 0; j < 16; ++j) tw[j] = f2bf(Win[h * INP + kmapA(lane, j)]);
    a_win[tt] = __builtin_bit_cast(v16bf, tw);
  }
  {
    v16us tmp;
#pragma unroll
    for (int j = 0; j < 16; ++j) {
      int h = 32 * wv + kmapA(lane, j);
      tmp[j] = (row < OUTN) ? f2bf(Wout[row * HID + h]) : (unsigned short)0;
    }
    a_wout = __builtin_bit_cast(v16bf, tmp);
  }
  __syncthreads();

  // ---- per-lane state registers in WMMA C/D layout ----
  // h = 32*wv + 16*tt + v + 8*hi, b = bbase + ncol
  float Iv[2][8], mv[2][8], sv[2][8], rv[2][8], tl[2][8];
#pragma unroll
  for (int tt = 0; tt < 2; ++tt)
#pragma unroll
    for (int v = 0; v < 8; ++v) {
      Iv[tt][v] = 0.f; mv[tt][v] = 0.f; sv[tt][v] = 0.f; rv[tt][v] = 0.f;
      tl[tt][v] = -1.f;
    }

  const float dt    = 0.002f;
  const float vthr  = 1.0f;
  const float tref  = 5.0f * dt;
  const float ls    = __expf(-dt / 0.01f);
  const float lm    = __expf(-dt / 0.02f);
  const float ld    = __expf(-dt / 0.03f);
  const float oml   = 1.0f - lm;
  const float rgain = dt / 0.03f;

#pragma unroll 1
  for (int t = 0; t < T_STEPS; ++t) {
    const float tnow = dt * (float)t;

    // --- phase A: u_t = poutT @ r_{t-1} (K split) ; y_{t-1} = Wout @ r_{t-1}
    {
      int mt = wv & 3;
      v8f accu = {0.f, 0.f, 0.f, 0.f, 0.f, 0.f, 0.f, 0.f};
#pragma unroll
      for (int kk = 0; kk < 4; ++kk) {
        int kt = (wv >> 2) * 4 + kk;
        accu = wmma_bf16(a_pout[kk], lds_b_tile(&r_t[kt * 512 + lane * 16]),
                         accu);
      }
      v8f accy = {0.f, 0.f, 0.f, 0.f, 0.f, 0.f, 0.f, 0.f};
      accy = wmma_bf16(a_wout, lds_b_tile(&r_t[wv * 512 + lane * 16]), accy);
#pragma unroll
      for (int v = 0; v < 8; ++v)
        atomicAdd(&u_f32[(16 * mt + v + 8 * hi) * 16 + ncol], accu[v]);
      if (hi == 0) {  // valid output rows o = v (0..7)
#pragma unroll
        for (int v = 0; v < 8; ++v) atomicAdd(&y_f32[v * 16 + ncol], accy[v]);
      }
    }
    __syncthreads();

    // --- phase B: repack u to bf16 B tiles; drain y_{t-1}
#pragma unroll
    for (int rep = 0; rep < 2; ++rep) {
      int s = tid + rep * NTHREADS;
      int kt = s >> 9, s2 = s & 511;
      int l2 = s2 & 31, j = s2 >> 5;
      int n = l2 & 15, k = kmapB(l2, j);
      u_t[kt * 512 + l2 * 16 + j] = f2bf(u_f32[(32 * kt + k) * 16 + n]);
    }
    if (t > 0 && tid < 128) {
      int n = tid & 15, o = tid >> 4;
      out[((size_t)(t - 1) * BATCH + bbase + n) * OUTN + o] = y_f32[o * 16 + n];
      y_f32[tid] = 0.f;
    }
    __syncthreads();

    // --- phase C: I-update matmuls + elementwise + restage r; prefetch x
    {
      v16bf bu0 = lds_b_tile(&u_t[lane * 16]);
      v16bf bu1 = lds_b_tile(&u_t[512 + lane * 16]);
      v16bf bx  = lds_b_tile(&xs[t & 1][lane * 16]);
#pragma unroll
      for (int tt = 0; tt < 2; ++tt) {
        v8f acc = {0.f, 0.f, 0.f, 0.f, 0.f, 0.f, 0.f, 0.f};
        acc = wmma_bf16(a_pre[tt][0], bu0, acc);
        acc = wmma_bf16(a_pre[tt][1], bu1, acc);
        acc = wmma_bf16(a_win[tt], bx, acc);
        v8us rr8;
#pragma unroll
        for (int v = 0; v < 8; ++v) {
          float In = ls * Iv[tt][v] + acc[v];
          float nr = (tnow > tl[tt][v] + tref) ? 1.f : 0.f;
          float mm = nr * (lm * mv[tt][v] + oml * In) * (1.f - sv[tt][v]);
          float rr = ld * rv[tt][v] + rgain * sv[tt][v];  // previous spike
          float sn = (mm > vthr) ? 1.f : 0.f;             // fast_sigmoid fwd
          float tn = tl[tt][v] + (tnow - tl[tt][v]) * sn;
          Iv[tt][v] = In; mv[tt][v] = mm; rv[tt][v] = rr;
          sv[tt][v] = sn; tl[tt][v] = tn;
          rr8[v] = f2bf(rr);
        }
        // contiguous 8 halves in B layout: l2 = ncol + 16*tt, j = 8*hi + v
        *reinterpret_cast<v8us*>(
            &r_t[wv * 512 + (ncol + 16 * tt) * 16 + 8 * hi]) = rr8;
      }
    }
    if (t + 1 < T_STEPS) {  // stage x for next step (double buffered)
      int l2 = tid & 31, j = tid >> 5;
      int n = l2 & 15, k = kmapB(l2, j);
      xs[(t + 1) & 1][l2 * 16 + j] =
          f2bf(x[((size_t)(t + 1) * BATCH + bbase + n) * INP + k]);
    }
    u_f32[tid] = 0.f;        // ready for next step's K-split reduce
    u_f32[tid + 512] = 0.f;
    __syncthreads();
  }

  // --- epilogue: y_{T-1} = Wout @ r_{T-1}
  {
    v8f accy = {0.f, 0.f, 0.f, 0.f, 0.f, 0.f, 0.f, 0.f};
    accy = wmma_bf16(a_wout, lds_b_tile(&r_t[wv * 512 + lane * 16]), accy);
    if (hi == 0) {
#pragma unroll
      for (int v = 0; v < 8; ++v) atomicAdd(&y_f32[v * 16 + ncol], accy[v]);
    }
  }
  __syncthreads();
  if (tid < 128) {
    int n = tid & 15, o = tid >> 4;
    out[((size_t)(T_STEPS - 1) * BATCH + bbase + n) * OUTN + o] =
        y_f32[o * 16 + n];
  }
}

extern "C" void kernel_launch(void* const* d_in, const int* in_sizes, int n_in,
                              void* d_out, int out_size, void* d_ws, size_t ws_size,
                              hipStream_t stream) {
  (void)in_sizes; (void)n_in; (void)out_size; (void)d_ws; (void)ws_size;
  const float* x    = (const float*)d_in[0];
  const float* Win  = (const float*)d_in[1];
  const float* Wout = (const float*)d_in[2];
  const float* pin  = (const float*)d_in[3];
  const float* pout = (const float*)d_in[4];
  const float* lvec = (const float*)d_in[5];
  float* out = (float*)d_out;

  dim3 grid(BATCH / NB);   // 8 workgroups, one per 16-batch group
  dim3 block(NTHREADS);    // 16 wave32s
  snn_lowrank_wmma_kernel<<<grid, block, 0, stream>>>(x, Win, Wout, pin, pout,
                                                      lvec, out);
}
